// DecoderRNN_22531398435110
// MI455X (gfx1250) — matrix operand order repaired and checked by reading, not compiled, not run on hardware
//
#include <hip/hip_runtime.h>
#include <hip/hip_bf16.h>

typedef __attribute__((ext_vector_type(16))) __bf16        v16bf;
typedef __attribute__((ext_vector_type(8)))  float         v8f;
typedef __attribute__((ext_vector_type(4)))  unsigned int  v4u;

#define HDIM   1024
#define BATCH  64
#define TSTEPS 32
#define VOCAB  32000
#define G3     3072            // 3*HDIM
#define ROWS   (TSTEPS*BATCH)  // 2048

static __device__ __forceinline__ unsigned short f2bf(float f) {
    unsigned int u = __float_as_uint(f);
    u += 0x7FFFu + ((u >> 16) & 1u);      // round-to-nearest-even
    return (unsigned short)(u >> 16);
}

union Frag { v16bf v; v4u u[2]; };

// A fragment: 16x32 bf16 tile at (row0,k0) of row-major [*,K] (ISA 7.12.2 layout)
// lanes 0-15: M=lane,    K = {k0..k0+7, k0+16..k0+23}
// lanes16-31: M=lane-16, K = {k0+8..k0+15, k0+24..k0+31}
static __device__ __forceinline__ Frag load_a(const unsigned short* __restrict__ A,
                                              int row0, int k0, int K, int lane) {
    int half = lane >> 4;
    const unsigned short* p = A + (size_t)(row0 + (lane & 15)) * K + k0 + half * 8;
    Frag f;
    f.u[0] = *(const v4u*)(p);
    f.u[1] = *(const v4u*)(p + 16);
    return f;
}

// B fragment: 32x16 tile of W^T where W is [N,K] row-major bf16
// lanes 0-15: N=n0+lane,    K = {k0..k0+7, k0+16..k0+23}
// lanes16-31: N=n0+lane-16, K = {k0+8..k0+15, k0+24..k0+31}
static __device__ __forceinline__ Frag load_b(const unsigned short* __restrict__ W,
                                              int n0, int k0, int K, int lane) {
    int half = lane >> 4;
    const unsigned short* p = W + (size_t)(n0 + (lane & 15)) * K + k0 + half * 8;
    Frag f;
    f.u[0] = *(const v4u*)(p);
    f.u[1] = *(const v4u*)(p + 16);
    return f;
}

static __device__ __forceinline__ v8f wmma_bf16(const Frag& a, const Frag& b, v8f c) {
    return __builtin_amdgcn_wmma_f32_16x16x32_bf16(false, a.v, false, b.v,
                                                   (short)0, c, false, false);
}

struct FragSet { Frag a0, a1, b0, b1, b2, b3; };

static __device__ __forceinline__ FragSet load_set(const unsigned short* __restrict__ A,
                                                   const unsigned short* __restrict__ W,
                                                   int m0, int n0, int k0, int K, int lane) {
    FragSet s;
    s.a0 = load_a(A, m0,      k0, K, lane);
    s.a1 = load_a(A, m0 + 16, k0, K, lane);
    s.b0 = load_b(W, n0,      k0, K, lane);
    s.b1 = load_b(W, n0 + 16, k0, K, lane);
    s.b2 = load_b(W, n0 + 32, k0, K, lane);
    s.b3 = load_b(W, n0 + 48, k0, K, lane);
    return s;
}

static __device__ __forceinline__ void compute_set(const FragSet& s, v8f acc[2][4]) {
    acc[0][0] = wmma_bf16(s.a0, s.b0, acc[0][0]);
    acc[1][0] = wmma_bf16(s.a1, s.b0, acc[1][0]);
    acc[0][1] = wmma_bf16(s.a0, s.b1, acc[0][1]);
    acc[1][1] = wmma_bf16(s.a1, s.b1, acc[1][1]);
    acc[0][2] = wmma_bf16(s.a0, s.b2, acc[0][2]);
    acc[1][2] = wmma_bf16(s.a1, s.b2, acc[1][2]);
    acc[0][3] = wmma_bf16(s.a0, s.b3, acc[0][3]);
    acc[1][3] = wmma_bf16(s.a1, s.b3, acc[1][3]);
}

// ---------------- elementwise helpers ----------------

__global__ void f32_to_bf16_kernel(const float* __restrict__ src,
                                   unsigned short* __restrict__ dst, size_t n) {
    size_t i = (size_t)blockIdx.x * blockDim.x + threadIdx.x;
    size_t stride = (size_t)gridDim.x * blockDim.x;
    for (; i < n; i += stride) dst[i] = f2bf(src[i]);
}

// tokens = [SOS, target[:, :T-1]]; x = relu(emb[tok]) -> bf16, row r = t*B+b
__global__ void embed_relu_bf16(const float* __restrict__ emb,
                                const int* __restrict__ target,
                                const int* __restrict__ sos,
                                unsigned short* __restrict__ xbf) {
    int r = blockIdx.x;                 // 0..2047
    int t = r >> 6, b = r & 63;
    int tok = (t == 0) ? sos[0] : target[b * TSTEPS + (t - 1)];
    const float* src = emb + (size_t)tok * HDIM;
    unsigned short* dst = xbf + (size_t)r * HDIM;
    for (int j = threadIdx.x; j < HDIM; j += blockDim.x) {
        float v = src[j];
        dst[j] = f2bf(v > 0.f ? v : 0.f);
    }
}

// ---------------- WMMA GEMM: D[M,N] = A[M,K](bf16) @ W[N,K]^T(bf16) + bias ----------------
// block = 128 threads (4 waves, 2x2), block tile 64(M) x 128(N), wave tile 32 x 64.
// Ping-pong double buffering (manual unroll x2): no rotation copies, loads of one set
// stay in flight across the 8 WMMAs of the other set.
// __launch_bounds__(128,1): allocate for 1 wave/EU so both fragment sets + 64 acc VGPRs
// stay in registers (no scratch spills).
// MODE 0: D row-major [M,N].  MODE 1: rows r=t*B+b scattered to logits layout [B,T,V].

template <int MODE>
__global__ __launch_bounds__(128, 1)
void gemm_bf16_wmma(const unsigned short* __restrict__ A,
                    const unsigned short* __restrict__ W,
                    const float* __restrict__ bias,
                    float* __restrict__ D,
                    int N, int K) {
    const int lane = threadIdx.x & 31;
    const int wave = threadIdx.x >> 5;
    const int m0 = blockIdx.y * 64 + (wave & 1) * 32;
    const int n0 = blockIdx.x * 128 + (wave >> 1) * 64;
    const int nl = lane & 15;

    v8f acc[2][4];
    #pragma unroll
    for (int j = 0; j < 4; ++j) {
        float bv = bias[n0 + j * 16 + nl];     // bias depends only on N (constant per lane)
        #pragma unroll
        for (int i = 0; i < 8; ++i) { acc[0][j][i] = bv; acc[1][j][i] = bv; }
    }

    FragSet s0 = load_set(A, W, m0, n0, 0, K, lane);
    FragSet s1;
    int k = 32;
    #pragma unroll 1
    for (int it = 0; it < K / 64 - 1; ++it) {   // K=1024 -> 15 iterations
        s1 = load_set(A, W, m0, n0, k, K, lane); k += 32;
        compute_set(s0, acc);
        s0 = load_set(A, W, m0, n0, k, K, lane); k += 32;
        compute_set(s1, acc);
    }
    s1 = load_set(A, W, m0, n0, k, K, lane);     // k = K-32
    compute_set(s0, acc);
    compute_set(s1, acc);

    // store: C/D layout is N=lane%16, M = VGPRidx + 8*(lane/16)
    const int half8 = (lane >> 4) << 3;
    #pragma unroll
    for (int mi = 0; mi < 2; ++mi) {
        int mb = m0 + mi * 16 + half8;
        #pragma unroll
        for (int j = 0; j < 4; ++j) {
            int n = n0 + j * 16 + nl;
            #pragma unroll
            for (int i = 0; i < 8; ++i) {
                if (MODE == 0) {
                    D[(size_t)(mb + i) * N + n] = acc[mi][j][i];
                } else {
                    int r = mb + i;                    // r = t*B + b
                    int bb = r & (BATCH - 1), t = r >> 6;
                    D[((size_t)bb * TSTEPS + t) * VOCAB + n] = acc[mi][j][i];
                }
            }
        }
    }
}

// ---------------- GRU gate fusion (fp32 master state) ----------------

__global__ void gru_gate(const float* __restrict__ gi_t,   // [B,3H] slice for step t
                         const float* __restrict__ gh,     // [B,3H]
                         float* __restrict__ h,            // [B,H] fp32 master
                         unsigned short* __restrict__ h_bf,
                         unsigned short* __restrict__ hs_bf_t) {
    int idx = blockIdx.x * blockDim.x + threadIdx.x;       // < 64*1024
    int b = idx >> 10, j = idx & 1023;
    const float* gib = gi_t + (size_t)b * G3;
    const float* ghb = gh   + (size_t)b * G3;
    float r = 1.f / (1.f + __expf(-(gib[j]          + ghb[j])));
    float z = 1.f / (1.f + __expf(-(gib[HDIM + j]   + ghb[HDIM + j])));
    float n = tanhf(gib[2 * HDIM + j] + r * ghb[2 * HDIM + j]);
    float hnew = (1.f - z) * n + z * h[idx];
    h[idx] = hnew;
    unsigned short bf = f2bf(hnew);
    h_bf[idx] = bf;
    hs_bf_t[idx] = bf;
}

// ---------------- log-softmax ----------------

__global__ void lse_kernel(const float* __restrict__ logits, float* __restrict__ stats) {
    int row = blockIdx.x;                       // row = b*T + t (d_out ordering)
    const float* x = logits + (size_t)row * VOCAB;
    __shared__ float red[256];
    float m = -1e30f;
    for (int j = threadIdx.x; j < VOCAB; j += 256) m = fmaxf(m, x[j]);
    red[threadIdx.x] = m; __syncthreads();
    for (int s = 128; s > 0; s >>= 1) {
        if (threadIdx.x < s) red[threadIdx.x] = fmaxf(red[threadIdx.x], red[threadIdx.x + s]);
        __syncthreads();
    }
    m = red[0]; __syncthreads();
    float sum = 0.f;
    for (int j = threadIdx.x; j < VOCAB; j += 256) sum += __expf(x[j] - m);
    red[threadIdx.x] = sum; __syncthreads();
    for (int s = 128; s > 0; s >>= 1) {
        if (threadIdx.x < s) red[threadIdx.x] += red[threadIdx.x + s];
        __syncthreads();
    }
    if (threadIdx.x == 0) stats[row] = m + __logf(red[0]);
}

__global__ void sub_lse(float* __restrict__ y, const float* __restrict__ stats) {
    int row = blockIdx.y;                            // 0..ROWS-1
    int col = blockIdx.x * 256 + threadIdx.x;        // VOCAB % 256 == 0
    y[(size_t)row * VOCAB + col] -= stats[row];
}

// ---------------- launch ----------------

extern "C" void kernel_launch(void* const* d_in, const int* in_sizes, int n_in,
                              void* d_out, int out_size, void* d_ws, size_t ws_size,
                              hipStream_t stream) {
    (void)in_sizes; (void)n_in; (void)out_size; (void)ws_size;
    const float* enc_hidden = (const float*)d_in[1];
    const int*   target     = (const int*)d_in[3];
    const int*   sos        = (const int*)d_in[4];
    const float* emb        = (const float*)d_in[6];
    const float* W_ih       = (const float*)d_in[7];
    const float* W_hh       = (const float*)d_in[8];
    const float* b_ih       = (const float*)d_in[9];
    const float* b_hh       = (const float*)d_in[10];
    const float* out_W      = (const float*)d_in[11];
    const float* out_b      = (const float*)d_in[12];
    float* out = (float*)d_out;

    // workspace carve-up (~113 MB total, all 16B aligned)
    char* ws = (char*)d_ws;
    unsigned short* Wih_bf  = (unsigned short*)ws; ws += (size_t)G3 * HDIM * 2;
    unsigned short* Whh_bf  = (unsigned short*)ws; ws += (size_t)G3 * HDIM * 2;
    unsigned short* outW_bf = (unsigned short*)ws; ws += (size_t)VOCAB * HDIM * 2;
    unsigned short* x_bf    = (unsigned short*)ws; ws += (size_t)ROWS * HDIM * 2;
    unsigned short* hs_bf   = (unsigned short*)ws; ws += (size_t)ROWS * HDIM * 2;
    unsigned short* h_bf    = (unsigned short*)ws; ws += (size_t)BATCH * HDIM * 2;
    float* gi    = (float*)ws; ws += (size_t)ROWS * G3 * 4;
    float* gh    = (float*)ws; ws += (size_t)BATCH * G3 * 4;
    float* h     = (float*)ws; ws += (size_t)BATCH * HDIM * 4;
    float* stats = (float*)ws; ws += (size_t)ROWS * 4;

    // 1) one-time bf16 conversions (bf16 out_W = 65 MB -> stays resident in 192 MB L2)
    f32_to_bf16_kernel<<<2048, 256, 0, stream>>>(W_ih,  Wih_bf,  (size_t)G3 * HDIM);
    f32_to_bf16_kernel<<<2048, 256, 0, stream>>>(W_hh,  Whh_bf,  (size_t)G3 * HDIM);
    f32_to_bf16_kernel<<<8192, 256, 0, stream>>>(out_W, outW_bf, (size_t)VOCAB * HDIM);
    f32_to_bf16_kernel<<<256,  256, 0, stream>>>(enc_hidden, h_bf, (size_t)BATCH * HDIM);
    hipMemcpyAsync(h, enc_hidden, (size_t)BATCH * HDIM * 4, hipMemcpyDeviceToDevice, stream);

    // 2) embedding + relu -> bf16
    embed_relu_bf16<<<ROWS, 256, 0, stream>>>(emb, target, sos, x_bf);

    // 3) gi = x @ W_ih^T + b_ih   [2048 x 3072]
    gemm_bf16_wmma<0><<<dim3(G3 / 128, ROWS / 64), 128, 0, stream>>>(
        x_bf, Wih_bf, b_ih, gi, G3, HDIM);

    // 4) sequential GRU scan (32 steps)
    for (int t = 0; t < TSTEPS; ++t) {
        gemm_bf16_wmma<0><<<dim3(G3 / 128, BATCH / 64), 128, 0, stream>>>(
            h_bf, Whh_bf, b_hh, gh, G3, HDIM);
        gru_gate<<<(BATCH * HDIM) / 256, 256, 0, stream>>>(
            gi + (size_t)t * BATCH * G3, gh, h, h_bf, hs_bf + (size_t)t * BATCH * HDIM);
    }

    // 5) logits = hs @ out_W^T + out_b, scattered to [B,T,V] in d_out
    gemm_bf16_wmma<1><<<dim3(VOCAB / 128, ROWS / 64), 128, 0, stream>>>(
        hs_bf, outW_bf, out_b, out, VOCAB, HDIM);

    // 6) in-place log-softmax over V
    lse_kernel<<<ROWS, 256, 0, stream>>>(out, stats);
    sub_lse<<<dim3(VOCAB / 256, ROWS), 256, 0, stream>>>(out, stats);

    // 7) h_final tail
    hipMemcpyAsync(out + (size_t)ROWS * VOCAB, h, (size_t)BATCH * HDIM * 4,
                   hipMemcpyDeviceToDevice, stream);
}
